// DelayLineRNN2xOS_84301618086391
// MI455X (gfx1250) — matrix operand order repaired and checked by reading, not compiled, not run on hardware
//
#include <hip/hip_runtime.h>
#include <math.h>

// ---------------- problem dims (fixed by the reference) ----------------
constexpr int Bn = 64;
constexpr int Tn = 1024;
constexpr int Dn = 1024;
constexpr int Hn = 1024;
constexpr long STATES_ELEMS = (long)Bn * Tn * Hn;   // 67,108,864

// ---------------- CDNA5 WMMA vector types ----------------
typedef __attribute__((ext_vector_type(16))) __bf16       v16bf;
typedef __attribute__((ext_vector_type(8)))  float        v8f;
typedef __attribute__((ext_vector_type(4)))  unsigned int v4u;
typedef __attribute__((ext_vector_type(8)))  unsigned int v8u;
typedef __attribute__((ext_vector_type(4)))  int          v4i;

// gfx1250 async global->LDS path (ASYNCcnt-tracked), with sync fallback
#if defined(__has_builtin)
#  if __has_builtin(__builtin_amdgcn_global_load_async_to_lds_b128) && \
      __has_builtin(__builtin_amdgcn_s_wait_asynccnt)
#    define HAVE_ASYNC_LDS 1
#  endif
#endif

__device__ __forceinline__ unsigned short f32_to_bf16(float f) {
  unsigned u = __float_as_uint(f);
  u += 0x7FFFu + ((u >> 16) & 1u);   // round-to-nearest-even
  return (unsigned short)(u >> 16);
}

__device__ __forceinline__ v8f wmma_bf16(v16bf a, v16bf b, v8f c) {
  return __builtin_amdgcn_wmma_f32_16x16x32_bf16(false, a, false, b,
                                                 (short)0, c, false, false);
}

__device__ __forceinline__ void async_lds_b128(const void* gsrc, void* ldst) {
#if HAVE_ASYNC_LDS
  // signature per hipcc diagnostic: (int4*, int4*, Imm offset, Imm cpol)
  __builtin_amdgcn_global_load_async_to_lds_b128(
      (v4i*)gsrc, (v4i*)ldst, 0, 0);
#else
  *(v4u*)ldst = *(const v4u*)gsrc;
#endif
}

// ---------------- init / conversion kernels ----------------
__global__ void zero_ws_kernel(unsigned* __restrict__ p, long n) {
  long i = (long)blockIdx.x * blockDim.x + threadIdx.x;
  long s = (long)gridDim.x * blockDim.x;
  for (; i < n; i += s) p[i] = 0u;
}

__global__ void cvt_bf16_kernel(const float* __restrict__ src,
                                unsigned short* __restrict__ dst, long n) {
  long i = (long)blockIdx.x * blockDim.x + threadIdx.x;
  long s = (long)gridDim.x * blockDim.x;
  for (; i < n; i += s) dst[i] = f32_to_bf16(src[i]);
}

// ---------------- Phase 1: gi = x @ W_ih^T + (b_ih + b_hh) ----------------
// WG = 8 waves as 4(M) x 2(N); each wave computes a 32x32 register block
// (4 accumulators, 4 WMMA per K-step from 2 A-frags x 2 B-frags).
// B operand (W_ih 64x32 K-slice) is staged in LDS via async-to-LDS,
// double buffered; A-frags stream from global (L2-resident bf16 x).
__global__ void __launch_bounds__(256) gemm_gi_kernel(
    const unsigned short* __restrict__ xb,      // [B*T, D] bf16
    const unsigned short* __restrict__ wih,     // [H, D] bf16
    const float* __restrict__ b_ih,
    const float* __restrict__ b_hh,
    float* __restrict__ out)                    // states region, [B*T, H]
{
  __shared__ unsigned short btile[2][64][32];   // 2 x 4KB double buffer

  const int tid   = threadIdx.x;
  const int lane  = tid & 31;
  const int wave  = tid >> 5;
  const int nGrp  = blockIdx.x & 15;            // 16 N-groups of 64 cols
  const int mGrp  = blockIdx.x >> 4;            // 512 M-groups of 128 rows
  const int mBase = mGrp * 128 + (wave >> 1) * 32;
  const int nLoc  = (wave & 1) * 32;
  const int nBase = nGrp * 64 + nLoc;
  const int lm    = lane & 15;
  const int half  = lane >> 4;

  // staging: each thread owns one 16B chunk of the 64x32 bf16 slice
  const int cr = tid >> 2;                      // row 0..63 (W_ih row)
  const int cc = (tid & 3) * 8;                 // 8-short chunk
  const unsigned short* gB = wih + (long)(nGrp * 64 + cr) * Dn + cc;
  unsigned short* lBp[2] = { &btile[0][cr][cc], &btile[1][cr][cc] };

  const unsigned short* aRow0 = xb + (long)(mBase + lm) * Dn;
  const unsigned short* aRow1 = aRow0 + (long)16 * Dn;

  v8f acc00 = {}, acc01 = {}, acc10 = {}, acc11 = {};

  async_lds_b128(gB, lBp[0]);
  for (int k0 = 0; k0 < Dn; k0 += 32) {
    const int cur = (k0 >> 5) & 1;
    if (k0 + 32 < Dn) async_lds_b128(gB + k0 + 32, lBp[cur ^ 1]);
#if HAVE_ASYNC_LDS
    if (k0 + 32 < Dn) __builtin_amdgcn_s_wait_asynccnt(1);  // current buf landed
    else              __builtin_amdgcn_s_wait_asynccnt(0);
#endif
    __syncthreads();                            // copies visible to all waves

    v8u ar0, ar1, br0, br1;
    ar0.lo = *(const v4u*)(aRow0 + k0 + half * 8);
    ar0.hi = *(const v4u*)(aRow0 + k0 + half * 8 + 16);
    ar1.lo = *(const v4u*)(aRow1 + k0 + half * 8);
    ar1.hi = *(const v4u*)(aRow1 + k0 + half * 8 + 16);
    const unsigned short* bp0 = &btile[cur][nLoc + lm][half * 8];
    const unsigned short* bp1 = &btile[cur][nLoc + 16 + lm][half * 8];
    br0.lo = *(const v4u*)bp0;  br0.hi = *(const v4u*)(bp0 + 16);
    br1.lo = *(const v4u*)bp1;  br1.hi = *(const v4u*)(bp1 + 16);

    const v16bf a0 = __builtin_bit_cast(v16bf, ar0);
    const v16bf a1 = __builtin_bit_cast(v16bf, ar1);
    const v16bf b0 = __builtin_bit_cast(v16bf, br0);
    const v16bf b1 = __builtin_bit_cast(v16bf, br1);
    acc00 = wmma_bf16(a0, b0, acc00);
    acc01 = wmma_bf16(a0, b1, acc01);
    acc10 = wmma_bf16(a1, b0, acc10);
    acc11 = wmma_bf16(a1, b1, acc11);

    __syncthreads();                            // reads done before buf reuse
  }

  const int n0 = nBase + lm;
  const int n1 = nBase + 16 + lm;
  const float bias0 = b_ih[n0] + b_hh[n0];
  const float bias1 = b_ih[n1] + b_hh[n1];
#pragma unroll
  for (int v = 0; v < 8; ++v) {
    const long r0 = (long)(mBase + half * 8 + v) * Hn;
    const long r1 = (long)(mBase + 16 + half * 8 + v) * Hn;
    out[r0 + n0] = acc00[v] + bias0;
    out[r0 + n1] = acc01[v] + bias1;
    out[r1 + n0] = acc10[v] + bias0;
    out[r1 + n1] = acc11[v] + bias1;
  }
}

// ---------------- Phase 2: persistent delay-line recurrence ----------------
// 64 WGs (one per 16-col W_hh slab in 32KB LDS), 8 waves = 2 chains x 4 M-tiles;
// h ping-pongs in a bf16 L2-resident buffer; one device barrier per step.
__global__ void __launch_bounds__(256, 1) rnn_steps_kernel(
    const unsigned short* __restrict__ whh,   // [H, H] bf16 row-major
    float* __restrict__ out,                  // states (holds gi) + h_final tail
    unsigned short* __restrict__ hbuf,        // [2 buf][2 chain][B][H] bf16
    unsigned* __restrict__ ctr)
{
  __shared__ v4u wslab[2048];                 // 16 rows x 1024 K bf16 = 32 KB

  const int tid   = threadIdx.x;
  const int nBase = blockIdx.x * 16;

  { // cooperative slab preload (async path when available)
    const unsigned short* src = whh + (long)nBase * Hn;
    for (int i = tid; i < 2048; i += 256)
      async_lds_b128(src + (long)i * 8, &wslab[i]);
#if HAVE_ASYNC_LDS
    __builtin_amdgcn_s_wait_asynccnt(0);
#endif
  }
  __syncthreads();

  const int lane  = tid & 31;
  const int wave  = tid >> 5;
  const int chain = wave & 1;                 // 0 = even line, 1 = odd line
  const int mBase = (wave >> 1) * 16;         // batch tile
  const int lm    = lane & 15;
  const int half  = lane >> 4;
  const int n     = nBase + lm;

  constexpr int HALF_T = Tn / 2;
  for (int step = 0; step < HALF_T; ++step) {
    const int cur = step & 1;
    const int t = (chain == 0) ? (2 * step)
                               : ((step == 0) ? (Tn - 1) : (2 * step - 1));
    const long colBase = (long)t * Hn + n;

    // acc <- gi slot (in-place consume of d_out)
    v8f acc;
#pragma unroll
    for (int v = 0; v < 8; ++v)
      acc[v] = out[(long)(mBase + half * 8 + v) * (long)(Tn * Hn) + colBase];

    // acc += h_prev @ W_hh^T
    const unsigned short* aRow =
        hbuf + ((long)cur * 2 + chain) * (Bn * Hn) + (long)(mBase + lm) * Hn;
    for (int k0 = 0; k0 < Hn; k0 += 32) {
      v8u ar, br;
      ar.lo = *(const v4u*)(aRow + k0 + half * 8);
      ar.hi = *(const v4u*)(aRow + k0 + half * 8 + 16);
      const int bi = lm * 128 + (k0 >> 3) + half;   // v4u index into slab
      br.lo = wslab[bi];
      br.hi = wslab[bi + 2];
      acc = wmma_bf16(__builtin_bit_cast(v16bf, ar),
                      __builtin_bit_cast(v16bf, br), acc);
    }

    // h = tanh(acc); write states (fp32) + next-step A operand (bf16)
    unsigned short* hn =
        hbuf + ((long)(cur ^ 1) * 2 + chain) * (Bn * Hn);
#pragma unroll
    for (int v = 0; v < 8; ++v) {
      const int brow = mBase + half * 8 + v;
      const float hv = tanhf(acc[v]);
      out[(long)brow * (long)(Tn * Hn) + colBase] = hv;
      hn[(long)brow * Hn + n] = f32_to_bf16(hv);
      if (chain == 0 && step == HALF_T - 1)              // h_final = even chain
        out[STATES_ELEMS + (long)brow * Hn + n] = hv;
    }

    // device-scope barrier between steps (monotonic counter)
    __syncthreads();
    if (tid == 0) {
      __threadfence();
      atomicAdd(ctr, 1u);
      const unsigned target = (unsigned)(step + 1) * gridDim.x;
      while (__hip_atomic_load(ctr, __ATOMIC_ACQUIRE,
                               __HIP_MEMORY_SCOPE_AGENT) < target) {
        __builtin_amdgcn_s_sleep(2);
      }
    }
    __syncthreads();
  }
}

// ---------------- host launcher ----------------
extern "C" void kernel_launch(void* const* d_in, const int* in_sizes, int n_in,
                              void* d_out, int out_size, void* d_ws, size_t ws_size,
                              hipStream_t stream) {
  const float* x    = (const float*)d_in[0];
  const float* W_ih = (const float*)d_in[1];
  const float* W_hh = (const float*)d_in[2];
  const float* b_ih = (const float*)d_in[3];
  const float* b_hh = (const float*)d_in[4];
  float* out = (float*)d_out;

  // workspace layout (bytes)
  char* ws = (char*)d_ws;
  const long OFF_CTR = 0;                       // 1 KB  (barrier counter)
  const long OFF_H   = 1024;                    // 512 KB (h ping-pong, bf16)
  const long OFF_X   = OFF_H + 2L * 2 * Bn * Hn * 2;
  const long OFF_WIH = OFF_X + (long)Bn * Tn * Dn * 2;
  const long OFF_WHH = OFF_WIH + (long)Hn * Dn * 2;

  unsigned*       ctr  = (unsigned*)(ws + OFF_CTR);
  unsigned short* hbuf = (unsigned short*)(ws + OFF_H);
  unsigned short* xb   = (unsigned short*)(ws + OFF_X);
  unsigned short* wihb = (unsigned short*)(ws + OFF_WIH);
  unsigned short* whhb = (unsigned short*)(ws + OFF_WHH);

  // deterministic per-launch init: counter + both h buffers = 0 (h0 = 0)
  zero_ws_kernel<<<256, 256, 0, stream>>>((unsigned*)ws, (OFF_X - OFF_CTR) / 4);

  // fp32 -> bf16 staging (bandwidth-trivial vs. the GEMMs)
  cvt_bf16_kernel<<<4096, 256, 0, stream>>>(x,    xb,   (long)Bn * Tn * Dn);
  cvt_bf16_kernel<<<512,  256, 0, stream>>>(W_ih, wihb, (long)Hn * Dn);
  cvt_bf16_kernel<<<512,  256, 0, stream>>>(W_hh, whhb, (long)Hn * Hn);

  // Phase 1: gi straight into the states region of d_out
  gemm_gi_kernel<<<(Bn * Tn / 128) * (Hn / 64), 256, 0, stream>>>(
      xb, wihb, b_ih, b_hh, out);

  // Phase 2: persistent recurrence, 64 WGs (one per W_hh slab)
  rnn_steps_kernel<<<Hn / 16, 256, 0, stream>>>(whhb, out, hbuf, ctr);
}
